// ProteinEGNN_28656021799366
// MI455X (gfx1250) — matrix operand order repaired and checked by reading, not compiled, not run on hardware
//
#include <hip/hip_runtime.h>
#include <hip/hip_bf16.h>
#include <math.h>

#define DD 128
#define RR 256
#define NBATCH 8
#define NEG_SLOPE 0.1f

typedef __attribute__((ext_vector_type(16))) __bf16          v16bf;
typedef __attribute__((ext_vector_type(8)))  unsigned int    v8u;
typedef __attribute__((ext_vector_type(8)))  float           v8f;

__device__ __forceinline__ float leaky(float x) { return fmaxf(x, NEG_SLOPE * x); }

// pack two f32 -> bf16x2 dword: round-half-up (+0x8000) then v_perm_b32 grabs
// the high halves.  a = {f1,f0} -> dword {bf16(f1), bf16(f0)}
__device__ __forceinline__ unsigned int pack_bf16(float f0, float f1) {
    unsigned int u0 = __float_as_uint(f0) + 0x8000u;
    unsigned int u1 = __float_as_uint(f1) + 0x8000u;
    return __builtin_amdgcn_perm(u1, u0, 0x07060302u);
}

__device__ __forceinline__ unsigned short f2bf(float f) {
    unsigned int u = __float_as_uint(f) + 0x8000u;
    return (unsigned short)(u >> 16);
}

// ---------------------------------------------------------------------------
// h = leaky(fresidues @ W_emb)   (2048 x 20) @ (20 x 128)
// ---------------------------------------------------------------------------
__global__ void emb_kernel(const float* __restrict__ fres,
                           const float* __restrict__ Wemb,
                           float* __restrict__ h) {
    int idx = blockIdx.x * 256 + threadIdx.x;
    int row = idx >> 7;
    int n   = idx & 127;
    float s = 0.f;
#pragma unroll
    for (int k = 0; k < 20; ++k) s += fres[row * 20 + k] * Wemb[k * DD + n];
    h[idx] = leaky(s);
}

// ---------------------------------------------------------------------------
// ti = h @ W_i,  tj = h @ W_j     (W_i = W_e1[:128], W_j = W_e1[128:256])
// ---------------------------------------------------------------------------
__global__ void titj_kernel(const float* __restrict__ h,
                            const float* __restrict__ We1,
                            float* __restrict__ ti,
                            float* __restrict__ tj) {
    int idx = blockIdx.x * 256 + threadIdx.x;
    int row = idx >> 7;
    int n   = idx & 127;
    float si = 0.f, sj = 0.f;
#pragma unroll 4
    for (int k = 0; k < DD; ++k) {
        float hv = h[row * DD + k];
        si += hv * We1[k * DD + n];
        sj += hv * We1[(DD + k) * DD + n];
    }
    ti[idx] = si;
    tj[idx] = sj;
}

// ---------------------------------------------------------------------------
// Fused edge MLP + gate + aggregation.
// One workgroup = (batch b, 8 consecutive i). One wave per i, loops all j.
// GEMM via v_wmma_f32_16x16x32_bf16: M=16 edge pairs, K=D, N=D.
// tj[b] staged to LDS with global_load_async_to_lds_b128 (ASYNCcnt path).
// ---------------------------------------------------------------------------
__global__ __launch_bounds__(256, 1) void edge_kernel(
    const float* __restrict__ ti, const float* __restrict__ tjg,
    const float* __restrict__ coords,
    const float* __restrict__ We1,     // w_d = We1 row 256
    const float* __restrict__ be1,
    const float* __restrict__ We2, const float* __restrict__ be2,
    const float* __restrict__ Winf, const float* __restrict__ binf_p,
    float* __restrict__ mpre_out) {

    extern __shared__ char smem[];
    float* tj_s     = (float*)smem;                 // 256 * 132 (padded stride)
    float* coords_s = tj_s + RR * 132;              // 256 * 4
    float* tib_s    = coords_s + RR * 4;            // 8 * 128  (ti + b_e1)
    float* wd_s     = tib_s + 8 * DD;               // 128
    unsigned short* w_s = (unsigned short*)(wd_s + DD); // 4kk * 8nt * 32lane * 16 bf16

    const int tid  = threadIdx.x;
    const int wid  = tid >> 5;
    const int lane = tid & 31;
    const int half = lane >> 4;
    const int col  = lane & 15;

    const int b      = blockIdx.x >> 5;        // 0..7
    const int i_base = (blockIdx.x & 31) * 8;
    const int i_idx  = i_base + wid;

    // ---- async stage tj[b] -> LDS (16B chunks; LDS row stride 528B = 33*16B)
    for (int c = 0; c < 32; ++c) {
        int ci = tid + 256 * c;                // chunk 0..8191
        int j  = ci >> 5;
        int kg = ci & 31;                      // 4-float group within row
        unsigned int lds_off = (unsigned int)(size_t)(tj_s + j * 132 + kg * 4);
        const float* gp = tjg + (size_t)(b * RR + j) * DD + kg * 4;
        asm volatile("global_load_async_to_lds_b128 %0, %1, off"
                     :: "v"(lds_off), "v"(gp) : "memory");
    }
    // ---- stage coordinates of all j
    for (int j = tid; j < RR; j += 256) {
        coords_s[j * 4 + 0] = coords[(b * RR + j) * 3 + 0];
        coords_s[j * 4 + 1] = coords[(b * RR + j) * 3 + 1];
        coords_s[j * 4 + 2] = coords[(b * RR + j) * 3 + 2];
    }
    // ---- stage (ti[i] + b_e1) for this wave's i
#pragma unroll
    for (int t = 0; t < 4; ++t) {
        int k = lane + t * 32;
        tib_s[wid * DD + k] = ti[(b * RR + i_idx) * DD + k] + be1[k];
    }
    // ---- stage w_d row
    if (tid < DD) wd_s[tid] = We1[2 * DD * DD + tid];
    // ---- stage W_e2 pre-swizzled into per-lane WMMA B-operand layout (bf16)
    //      chunk (kk, nt, lane): slot s -> K = kk*32 + 8*half + (s < 8 ? s : s + 8), N = nt*16 + col
    for (int idx = tid; idx < 1024; idx += 256) {
        int kk = idx >> 8, rem = idx & 255;
        int nt = rem >> 5, ln = rem & 31;
        int nn = nt * 16 + (ln & 15);
        int kb = kk * 32 + 8 * (ln >> 4);
#pragma unroll
        for (int s = 0; s < 16; ++s)
            w_s[idx * 16 + s] = f2bf(We2[(kb + (s < 8 ? s : s + 8)) * DD + nn]);
    }
    asm volatile("s_wait_asynccnt 0" ::: "memory");
    __syncthreads();

    // per-lane broadcast constants
    float winf_l[8], be2_l[8];
#pragma unroll
    for (int nt = 0; nt < 8; ++nt) {
        winf_l[nt] = Winf[nt * 16 + col];
        be2_l[nt]  = be2[nt * 16 + col];
    }
    const float binf = binf_p[0];
    const float cx = coords[(b * RR + i_idx) * 3 + 0];
    const float cy = coords[(b * RR + i_idx) * 3 + 1];
    const float cz = coords[(b * RR + i_idx) * 3 + 2];

    float mpre[8];
#pragma unroll
    for (int nt = 0; nt < 8; ++nt) mpre[nt] = 0.f;

    const v8u* wmat = (const v8u*)w_s;
    const v8f vzero = {0.f, 0.f, 0.f, 0.f, 0.f, 0.f, 0.f, 0.f};

    for (int jt = 0; jt < 16; ++jt) {
        const int j = jt * 16 + col;      // A-matrix row (edge pair) for this lane
        const float dx = cx - coords_s[j * 4 + 0];
        const float dy = cy - coords_s[j * 4 + 1];
        const float dz = cz - coords_s[j * 4 + 2];
        const float d2 = dx * dx + dy * dy + dz * dz;

        v8f acc[8];
#pragma unroll
        for (int nt = 0; nt < 8; ++nt) acc[nt] = vzero;

#pragma unroll
        for (int kk = 0; kk < 4; ++kk) {
            // ---- build one A operand: leaky(ti[i,k] + tj[j,k] + d2*w_d[k] + b_e1[k])
            //      slot pair (2t,2t+1) -> consecutive k
            v8u au;
#pragma unroll
            for (int t = 0; t < 8; ++t) {
                const int sb = 2 * t;
                const int k  = kk * 32 + (sb < 8 ? sb + 8 * half : sb + 8 + 8 * half);
                float f0 = tib_s[wid * DD + k]     + tj_s[j * 132 + k]     + d2 * wd_s[k];
                float f1 = tib_s[wid * DD + k + 1] + tj_s[j * 132 + k + 1] + d2 * wd_s[k + 1];
                au[t] = pack_bf16(leaky(f0), leaky(f1));
            }
            const v16bf a = __builtin_bit_cast(v16bf, au);
#pragma unroll
            for (int nt = 0; nt < 8; ++nt) {
                v8u bu = wmat[(kk * 8 + nt) * 32 + lane];
                acc[nt] = __builtin_amdgcn_wmma_f32_16x16x32_bf16(
                    false, a, false, __builtin_bit_cast(v16bf, bu),
                    (short)0, acc[nt], false, false);
            }
        }

        // ---- epilogue: m = leaky(acc + b_e2)
#pragma unroll
        for (int nt = 0; nt < 8; ++nt)
#pragma unroll
            for (int r = 0; r < 8; ++r)
                acc[nt][r] = leaky(acc[nt][r] + be2_l[nt]);

        // gate e = sigmoid(m @ W_inf + b_inf) * (j != i); accumulate m_pre
#pragma unroll
        for (int r = 0; r < 8; ++r) {
            float p = 0.f;
#pragma unroll
            for (int nt = 0; nt < 8; ++nt) p += acc[nt][r] * winf_l[nt];
            p += __shfl_xor(p, 1, 32);
            p += __shfl_xor(p, 2, 32);
            p += __shfl_xor(p, 4, 32);
            p += __shfl_xor(p, 8, 32);
            const int jr = jt * 16 + r + 8 * half;   // global j of this C-row
            const float e = (jr == i_idx) ? 0.f : 1.f / (1.f + __expf(-(p + binf)));
#pragma unroll
            for (int nt = 0; nt < 8; ++nt) mpre[nt] += acc[nt][r] * e;
        }
    }

    // combine the two half-wave row groups, then store m_pre[i,:]
#pragma unroll
    for (int nt = 0; nt < 8; ++nt) mpre[nt] += __shfl_xor(mpre[nt], 16, 32);
    if (lane < 16) {
#pragma unroll
        for (int nt = 0; nt < 8; ++nt)
            mpre_out[(b * RR + i_idx) * DD + nt * 16 + col] = mpre[nt];
    }
}

// ---------------------------------------------------------------------------
// Node update: h = leaky([h, m_pre] @ W_h1 + b_h1) @ W_h2 + b_h2 + h  (in place)
// ---------------------------------------------------------------------------
__global__ __launch_bounds__(128) void node_kernel(
    float* __restrict__ h, const float* __restrict__ mpre,
    const float* __restrict__ Wh1, const float* __restrict__ bh1,
    const float* __restrict__ Wh2, const float* __restrict__ bh2,
    float* __restrict__ out, int write_out) {
    __shared__ float t_s[DD];
    const int row = blockIdx.x;
    const int n   = threadIdx.x;

    float s = bh1[n];
#pragma unroll 4
    for (int k = 0; k < DD; ++k) s += h[row * DD + k] * Wh1[k * DD + n];
#pragma unroll 4
    for (int k = 0; k < DD; ++k) s += mpre[row * DD + k] * Wh1[(DD + k) * DD + n];
    t_s[n] = leaky(s);
    __syncthreads();

    float s2 = bh2[n];
#pragma unroll 4
    for (int k = 0; k < DD; ++k) s2 += t_s[k] * Wh2[k * DD + n];
    const float hn = s2 + h[row * DD + n];
    h[row * DD + n] = hn;
    if (write_out) out[row * DD + n] = hn;
}

// ---------------------------------------------------------------------------
extern "C" void kernel_launch(void* const* d_in, const int* in_sizes, int n_in,
                              void* d_out, int out_size, void* d_ws, size_t ws_size,
                              hipStream_t stream) {
    const float* fres   = (const float*)d_in[0];   // (8,256,20)
    const float* coords = (const float*)d_in[1];   // (8,256,3)
    const float* Wemb   = (const float*)d_in[2];   // (20,128)
    const float* We1    = (const float*)d_in[3];   // (257,128)
    const float* be1    = (const float*)d_in[4];   // (128,)
    const float* We2    = (const float*)d_in[5];   // (128,128)
    const float* be2    = (const float*)d_in[6];   // (128,)
    const float* Winf   = (const float*)d_in[7];   // (128,1)
    const float* binf   = (const float*)d_in[8];   // (1,)
    const float* Wh1    = (const float*)d_in[9];   // (256,128)
    const float* bh1    = (const float*)d_in[10];  // (128,)
    const float* Wh2    = (const float*)d_in[11];  // (128,128)
    const float* bh2    = (const float*)d_in[12];  // (128,)

    const int NROW = NBATCH * RR;          // 2048
    const int NEL  = NROW * DD;            // 262144 floats per tensor

    float* h  = (float*)d_ws;
    float* ti = h + NEL;
    float* tj = ti + NEL;
    float* mp = tj + NEL;

    const size_t edge_lds =
        (size_t)(RR * 132 + RR * 4 + 8 * DD + DD) * sizeof(float) +
        (size_t)(4 * 8 * 32 * 16) * sizeof(unsigned short);   // = 176,640 B

    emb_kernel<<<NEL / 256, 256, 0, stream>>>(fres, Wemb, h);

    for (int it = 0; it < 2; ++it) {
        titj_kernel<<<NEL / 256, 256, 0, stream>>>(h, We1, ti, tj);
        edge_kernel<<<NBATCH * (RR / 8), 256, edge_lds, stream>>>(
            ti, tj, coords, We1, be1, We2, be2, Winf, binf, mp);
        node_kernel<<<NROW, 128, 0, stream>>>(
            h, mp, Wh1, bh1, Wh2, bh2, (float*)d_out, it == 1 ? 1 : 0);
    }
}
